// Attention_44470091383531
// MI455X (gfx1250) — matrix operand order repaired and checked by reading, not compiled, not run on hardware
//
#include <hip/hip_runtime.h>
#include <cstdint>

#define B_ 2
#define T_ 2048
#define C_ 2048
#define H_ 16
#define D_ 128
#define SCALE_ 0.08838834764831845f   // 1/sqrt(128)

typedef __attribute__((ext_vector_type(16))) __bf16 v16bf;
typedef __attribute__((ext_vector_type(8)))  float  v8f;
typedef __attribute__((ext_vector_type(4)))  unsigned int v4u;
typedef __attribute__((ext_vector_type(8)))  int v8i;
typedef __attribute__((ext_vector_type(4)))  int v4i;

#if __has_builtin(__builtin_amdgcn_tensor_load_to_lds)
#define HAVE_TDM 1
#else
#define HAVE_TDM 0
#endif

__device__ __forceinline__ __bf16 f2bf(float f) {
  unsigned u = __float_as_uint(f);
  u += 0x7FFFu + ((u >> 16) & 1u);          // round-to-nearest-even
  unsigned short h = (unsigned short)(u >> 16);
  return __builtin_bit_cast(__bf16, h);
}
__device__ __forceinline__ float bf2f(__bf16 b) {
  unsigned short h = __builtin_bit_cast(unsigned short, b);
  return __uint_as_float(((unsigned)h) << 16);
}
__device__ __forceinline__ v8f zero8() {
  v8f z;
#pragma unroll
  for (int i = 0; i < 8; ++i) z[i] = 0.0f;
  return z;
}
// A-matrix fragment (16x32 bf16): lane-half K chunks {kb..kb+7} and {kb+16..kb+23}
__device__ __forceinline__ v16bf load_fragA(const __bf16* p) {
  union { v16bf v; uint4 q[2]; } u;
  u.q[0] = *(const uint4*)(p);
  u.q[1] = *(const uint4*)(p + 16);
  return u.v;
}
// B-matrix fragment (32x16 bf16): lane-half K chunks {kb..kb+7} and {kb+16..kb+23}
__device__ __forceinline__ v16bf load_fragB(const __bf16* p) {
  union { v16bf v; uint4 q[2]; } u;
  u.q[0] = *(const uint4*)(p);
  u.q[1] = *(const uint4*)(p + 16);
  return u.v;
}

#if HAVE_TDM
// Issue a TDM 2D tile load: tile_d0 x tile_d1 (elements, 2B each) from a
// row-major tensor with row stride `stride_el` into LDS at byte addr lds_addr.
// D# bitfields per CDNA5 ISA ch.8 (group0: count/lds/global/type,
// group1: data_size/dims/tile/strides). Tensor dims set huge => no OOB clamp.
// Toolchain builtin is the 6-arg form:
//   (v4u g0, v8i g1, v4i g2, v4i g3, v8i g4, i32 cpol)
__device__ __forceinline__ void tdm_load_2d(unsigned lds_addr, const void* gptr,
                                            unsigned tile_d0, unsigned tile_d1,
                                            unsigned long long stride_el) {
  unsigned long long ga = (unsigned long long)(uintptr_t)gptr;
  const unsigned TD0 = 0x40000000u;   // tensor_dim0 (huge)
  const unsigned TD1 = 0x00100000u;   // tensor_dim1 (huge)
  v4u g0;
  g0[0] = 1u;                                          // count=1, user desc
  g0[1] = lds_addr;                                    // LDS byte address
  g0[2] = (unsigned)(ga & 0xFFFFFFFFu);                // global_addr[31:0]
  g0[3] = (unsigned)((ga >> 32) & 0x01FFFFFFu) | 0x80000000u;  // [56:32] | type=2
  v8i g1;
  g1[0] = (int)(1u << 16);                             // data_size=1 (2 bytes)
  g1[1] = (int)((TD0 & 0xFFFFu) << 16);                // td0[15:0]
  g1[2] = (int)(((TD1 & 0xFFFFu) << 16) | (TD0 >> 16));
  g1[3] = (int)((tile_d0 << 16) | (TD1 >> 16));
  g1[4] = (int)(tile_d1 & 0xFFFFu);                    // tile_dim1, tile_dim2=0
  g1[5] = (int)(unsigned)(stride_el & 0xFFFFFFFFu);    // dim0 stride
  g1[6] = (int)(unsigned)(stride_el >> 32);
  g1[7] = 0;
  v4i gz;
  gz[0] = 0; gz[1] = 0; gz[2] = 0; gz[3] = 0;
  v8i gz8;
#pragma unroll
  for (int i = 0; i < 8; ++i) gz8[i] = 0;
  __builtin_amdgcn_tensor_load_to_lds(g0, g1, gz, gz, gz8, 0);
}
#endif

// ---------------------------------------------------------------- fp32->bf16
__global__ void cvt_bf16_kernel(const float* __restrict__ src,
                                __bf16* __restrict__ dst, int n) {
  int i = blockIdx.x * blockDim.x + threadIdx.x;
  if (i < n) dst[i] = f2bf(src[i]);
}

// -------------------------------------------------------------------- GEMM
// Y = X[BT x C] * W^T. Block = 8 waves sharing one 16-row A tile staged in
// LDS by the Tensor Data Mover (double-buffered, TENSORcnt-pipelined).
// Each wave owns a 16x64 output tile (4 wmma columns).
// PROJ=true: bf16 out scattered to [B,H,T,D]; else fp32 row-major to out.
#define KCHUNK 256
#define STAGE_EL (16 * KCHUNK)

template <bool PROJ>
__global__ __launch_bounds__(256) void gemm_kernel(
    const __bf16* __restrict__ X, const __bf16* __restrict__ W, void* dstv) {
  __shared__ __align__(16) __bf16 sA[2][STAGE_EL];   // 16 KB
  const int lane = threadIdx.x & 31;
  const int wid  = threadIdx.x >> 5;
  const int mt = blockIdx.x >> 2;            // 256 M tiles of 16 rows
  const int ng = blockIdx.x & 3;             // 4 N groups of 512 cols
  const int nt = ng * 8 + wid;               // this wave's 64-col tile
  const int ln = lane & 15;
  const int hi = lane >> 4;

  const __bf16* Abase = X + (size_t)(mt * 16) * C_;

#if HAVE_TDM
  if (wid == 0) {
    unsigned lds0 = (unsigned)(uintptr_t)(void*)&sA[0][0];
    tdm_load_2d(lds0, Abase, KCHUNK, 16, C_);
  }
#endif

  v8f acc[4] = { zero8(), zero8(), zero8(), zero8() };

  const int NCHUNK = C_ / KCHUNK;            // 8
  for (int c = 0; c < NCHUNK; ++c) {
#if HAVE_TDM
    if (wid == 0) {
      if (c + 1 < NCHUNK) {
        unsigned lds = (unsigned)(uintptr_t)(void*)&sA[(c + 1) & 1][0];
        tdm_load_2d(lds, Abase + (c + 1) * KCHUNK, KCHUNK, 16, C_);
        __builtin_amdgcn_s_wait_tensorcnt((short)1);   // chunk c landed
      } else {
        __builtin_amdgcn_s_wait_tensorcnt((short)0);
      }
    }
#else
    {   // cooperative fallback staging: 256 threads x 2 x 16B
      const __bf16* src = Abase + c * KCHUNK;
      __bf16* dstl = &sA[c & 1][0];
#pragma unroll
      for (int e = 0; e < 2; ++e) {
        const int idx = ((int)threadIdx.x * 2 + e) * 8;  // element in tile
        const int rr = idx >> 8, rc = idx & (KCHUNK - 1);
        *(uint4*)&dstl[rr * KCHUNK + rc] =
            *(const uint4*)&src[(size_t)rr * C_ + rc];
      }
    }
#endif
    __syncthreads();
    const __bf16* abuf = &sA[c & 1][0];
#pragma unroll
    for (int kk = 0; kk < KCHUNK / 32; ++kk) {
      v16bf a = load_fragA(abuf + ln * KCHUNK + kk * 32 + (hi << 3));
      const int kglob = c * KCHUNK + kk * 32;
#pragma unroll
      for (int s = 0; s < 4; ++s) {
        const __bf16* bp =
            W + (size_t)(nt * 64 + s * 16 + ln) * C_ + kglob + (hi << 3);
        v16bf b = load_fragB(bp);
        acc[s] = __builtin_amdgcn_wmma_f32_16x16x32_bf16(
            false, a, false, b, (short)0, acc[s], false, false);
      }
    }
    __syncthreads();
  }

#pragma unroll
  for (int s = 0; s < 4; ++s) {
    const int col = nt * 64 + s * 16 + ln;
#pragma unroll
    for (int r = 0; r < 8; ++r) {
      const int row = mt * 16 + r + (hi << 3);
      if constexpr (PROJ) {
        const int h = col >> 7, d = col & (D_ - 1);
        const int b = row >> 11, t = row & (T_ - 1);
        ((__bf16*)dstv)[(((size_t)(b * H_ + h)) * T_ + t) * D_ + d] =
            f2bf(acc[s][r]);
      } else {
        ((float*)dstv)[(size_t)row * C_ + col] = acc[s][r];
      }
    }
  }
}

// ---------------------------------------------------------------- RoPE
// y[i]    = x[i]   *cos(f_i) - x[2i+1]*sin(f_i)   for i < 64
// y[i+64] = x[i+64]*cos(f_i) + x[2i]  *sin(f_i)
__global__ void rope_kernel(const __bf16* __restrict__ src,
                            __bf16* __restrict__ dst) {
  const int gid = blockIdx.x * blockDim.x + threadIdx.x;  // B*H*T*64 threads
  const int i = gid & 63;
  const int row = gid >> 6;            // (b*H + h)*T + t
  const int t = row & (T_ - 1);
  const __bf16* x = src + (size_t)row * D_;
  __bf16* y = dst + (size_t)row * D_;
  const float inv = __expf(-((2.0f * (float)i) / (float)D_) * __logf(10000.0f));
  const float ang = (float)t * inv;
  float s, c;
  __sincosf(ang, &s, &c);
  const float xi   = bf2f(x[i]);
  const float x2i1 = bf2f(x[2 * i + 1]);
  const float xh   = bf2f(x[i + 64]);
  const float x2i  = bf2f(x[2 * i]);
  y[i]      = f2bf(xi * c - x2i1 * s);
  y[i + 64] = f2bf(xh * c + x2i * s);
}

// ---------------------------------------------------------------- attention
__global__ __launch_bounds__(128) void flash_attn_kernel(
    const __bf16* __restrict__ Q, const __bf16* __restrict__ K,
    const __bf16* __restrict__ V, __bf16* __restrict__ O) {
  __shared__ __align__(16) __bf16 sP[4][16 * 32];    // P tiles, per wave
  __shared__ __align__(16) __bf16 sV[4][128 * 32];   // V^T tiles, per wave
  const int lane = threadIdx.x & 31;
  const int wid = threadIdx.x >> 5;
  const int ln = lane & 15, hi = lane >> 4;
  __bf16* sPw = sP[wid];
  __bf16* sVw = sV[wid];

  const int qidx = blockIdx.x * 4 + wid;     // 4096 q-tiles
  const int qt = qidx & 127;                 // T/16 tiles per head
  const int bh = qidx >> 7;
  const int q0 = qt * 16;
  const __bf16* Qh = Q + (size_t)bh * T_ * D_;
  const __bf16* Kh = K + (size_t)bh * T_ * D_;
  const __bf16* Vh = V + (size_t)bh * T_ * D_;

  v16bf aQ[4];
#pragma unroll
  for (int kk = 0; kk < 4; ++kk)
    aQ[kk] = load_fragA(Qh + (size_t)(q0 + ln) * D_ + kk * 32 + (hi << 3));

  v8f accO[8];
#pragma unroll
  for (int dt = 0; dt < 8; ++dt) accO[dt] = zero8();
  float m[8], l[8];
#pragma unroll
  for (int r = 0; r < 8; ++r) { m[r] = -1e30f; l[r] = 0.0f; }

  const int kend = q0 + 16;
  for (int k0 = 0; k0 < kend; k0 += 32) {
    // stage V tile transposed into LDS: sVw[d*32 + key]
    {
      const __bf16* vrow = Vh + (size_t)(k0 + lane) * D_;
#pragma unroll
      for (int c8 = 0; c8 < 16; ++c8) {
        union { uint4 q; unsigned short s[8]; } u;
        u.q = *(const uint4*)(vrow + c8 * 8);
#pragma unroll
        for (int e = 0; e < 8; ++e)
          sVw[(c8 * 8 + e) * 32 + lane] = __builtin_bit_cast(__bf16, u.s[e]);
      }
    }
    // S = Q * K^T   (two 16-key column blocks)
    v8f sc[2] = { zero8(), zero8() };
#pragma unroll
    for (int kk = 0; kk < 4; ++kk) {
#pragma unroll
      for (int sub = 0; sub < 2; ++sub) {
        const __bf16* kp =
            Kh + (size_t)(k0 + sub * 16 + ln) * D_ + kk * 32 + (hi << 3);
        v16bf bk = load_fragB(kp);
        sc[sub] = __builtin_amdgcn_wmma_f32_16x16x32_bf16(
            false, aQ[kk], false, bk, (short)0, sc[sub], false, false);
      }
    }
    // online softmax (rows live in lane-halves; reduce across 16 lanes)
#pragma unroll
    for (int r = 0; r < 8; ++r) {
      const int row = q0 + r + (hi << 3);
      float v0 = sc[0][r] * SCALE_;
      float v1 = sc[1][r] * SCALE_;
      if (k0 + ln > row) v0 = -1e30f;
      if (k0 + 16 + ln > row) v1 = -1e30f;
      float mx = fmaxf(v0, v1);
      mx = fmaxf(mx, __shfl_xor(mx, 1, 16));
      mx = fmaxf(mx, __shfl_xor(mx, 2, 16));
      mx = fmaxf(mx, __shfl_xor(mx, 4, 16));
      mx = fmaxf(mx, __shfl_xor(mx, 8, 16));
      const float mn = fmaxf(m[r], mx);
      const float alpha = __expf(m[r] - mn);
      const float p0 = __expf(v0 - mn);
      const float p1 = __expf(v1 - mn);
      float rs = p0 + p1;
      rs += __shfl_xor(rs, 1, 16);
      rs += __shfl_xor(rs, 2, 16);
      rs += __shfl_xor(rs, 4, 16);
      rs += __shfl_xor(rs, 8, 16);
      l[r] = l[r] * alpha + rs;
      m[r] = mn;
#pragma unroll
      for (int dt = 0; dt < 8; ++dt) accO[dt][r] *= alpha;
      const int lr = r + (hi << 3);
      sPw[lr * 32 + ln] = f2bf(p0);
      sPw[lr * 32 + 16 + ln] = f2bf(p1);
    }
    asm volatile("s_wait_dscnt 0x0" ::: "memory");
    // O += P * V
    v16bf aP = load_fragA(sPw + ln * 32 + (hi << 3));
#pragma unroll
    for (int dt = 0; dt < 8; ++dt) {
      v16bf bv = load_fragB(sVw + (dt * 16 + ln) * 32 + (hi << 3));
      accO[dt] = __builtin_amdgcn_wmma_f32_16x16x32_bf16(
          false, aP, false, bv, (short)0, accO[dt], false, false);
    }
  }
  // epilogue: normalize, write [B, T, H*D]
  const int b = bh >> 4, h = bh & 15;
#pragma unroll
  for (int r = 0; r < 8; ++r) {
    const int t = q0 + r + (hi << 3);
    const float inv = 1.0f / l[r];
    __bf16* orow = O + ((size_t)(b * T_ + t)) * C_ + h * D_;
#pragma unroll
    for (int dt = 0; dt < 8; ++dt)
      orow[dt * 16 + ln] = f2bf(accO[dt][r] * inv);
  }
}

// ---------------------------------------------------------------- launch
extern "C" void kernel_launch(void* const* d_in, const int* in_sizes, int n_in,
                              void* d_out, int out_size, void* d_ws, size_t ws_size,
                              hipStream_t stream) {
  const float* hs = (const float*)d_in[0];
  const float* qw = (const float*)d_in[1];
  const float* kw = (const float*)d_in[2];
  const float* vw = (const float*)d_in[3];
  const float* ow = (const float*)d_in[4];

  const size_t NX = (size_t)B_ * T_ * C_;   // 8,388,608
  const size_t NW = (size_t)C_ * C_;        // 4,194,304
  __bf16* Xb   = (__bf16*)d_ws;
  __bf16* Wqb  = Xb + NX;
  __bf16* Wkb  = Wqb + NW;
  __bf16* Wvb  = Wkb + NW;
  __bf16* Wob  = Wvb + NW;
  __bf16* Qraw = Wob + NW;
  __bf16* Kraw = Qraw + NX;
  __bf16* Vb   = Kraw + NX;
  __bf16* Qr   = Vb + NX;
  __bf16* Kr   = Qr + NX;
  __bf16* Ab   = Kr + NX;   // 144 MB total bf16 workspace

  cvt_bf16_kernel<<<(int)((NX + 255) / 256), 256, 0, stream>>>(hs, Xb, (int)NX);
  cvt_bf16_kernel<<<(int)((NW + 255) / 256), 256, 0, stream>>>(qw, Wqb, (int)NW);
  cvt_bf16_kernel<<<(int)((NW + 255) / 256), 256, 0, stream>>>(kw, Wkb, (int)NW);
  cvt_bf16_kernel<<<(int)((NW + 255) / 256), 256, 0, stream>>>(vw, Wvb, (int)NW);
  cvt_bf16_kernel<<<(int)((NW + 255) / 256), 256, 0, stream>>>(ow, Wob, (int)NW);

  gemm_kernel<true><<<1024, 256, 0, stream>>>(Xb, Wqb, (void*)Qraw);
  gemm_kernel<true><<<1024, 256, 0, stream>>>(Xb, Wkb, (void*)Kraw);
  gemm_kernel<true><<<1024, 256, 0, stream>>>(Xb, Wvb, (void*)Vb);

  const int ropeThreads = B_ * H_ * T_ * 64;  // 4,194,304
  rope_kernel<<<ropeThreads / 256, 256, 0, stream>>>(Qraw, Qr);
  rope_kernel<<<ropeThreads / 256, 256, 0, stream>>>(Kraw, Kr);

  flash_attn_kernel<<<1024, 128, 0, stream>>>(Qr, Kr, Vb, Ab);

  gemm_kernel<false><<<1024, 256, 0, stream>>>(Ab, Wob, d_out);
}
